// Conv1DStft_68582037782681
// MI455X (gfx1250) — matrix operand order repaired and checked by reading, not compiled, not run on hardware
//
#include <hip/hip_runtime.h>
#include <hip/hip_bf16.h>

// ---- problem constants (from reference) ----
#define T_LEN     480000
#define NFFT      1024
#define HOP       256
#define N_FRAMES  1876          // (T + 2*512 - 1024)/256 + 1
#define N_COLS    1026          // 513 freqs * {re, im}
#define M_ROWS    30016         // 16 * 1876
#define NT_PAD    72            // column tiles padded (72*16 = 1152 >= 1026), zero-filled
#define KB        32            // K-blocks of 32 (1024/32)
#define KCHUNK    256           // K staged per LDS chunk
#define KB_CHUNK  8             // K-blocks per chunk
#define N_CHUNKS  4
#define SROW      264           // LDS row stride in halves (256+8): dword stride 132 % 64 == 4 -> conflict-free
#define M_PER_BLK 32            // 2 M-tiles per block -> B traffic amortized 2x
#define N_BLOCKS  (M_ROWS / M_PER_BLK)   // 938

typedef __attribute__((ext_vector_type(16))) __bf16         v16bf;
typedef __attribute__((ext_vector_type(8)))  float          v8f;
typedef __attribute__((ext_vector_type(8)))  unsigned short ush8;

union V16U { v16bf v; ush8 u[2]; };

static __device__ __forceinline__ unsigned short f2bf(float f) {
    __bf16 h = (__bf16)f;                       // RNE
    return __builtin_bit_cast(unsigned short, h);
}
static __device__ __forceinline__ float bf2f(unsigned short u) {
    return (float)__builtin_bit_cast(__bf16, u);
}

// -----------------------------------------------------------------------------
// Kernel 1: build windowed DFT basis directly in WMMA B-fragment layout.
// B fragment (bf16 32x16, K x N) per lane L: N = L&15, K = j + 16*(L>>4), j=0..15.
// Storage per (ntile, kblock, lane): 16 bf16 hi then 16 bf16 lo (64 bytes).
// Columns c >= 1026 are zero padding (no guards needed in the GEMM hot loop).
// -----------------------------------------------------------------------------
__global__ __launch_bounds__(256) void stft_build_basis(
        const float* __restrict__ win, unsigned short* __restrict__ Bws) {
    int gid  = blockIdx.x * 256 + threadIdx.x;  // 72*32*32 = 73728 total
    int lane = gid & 31;
    int kb   = (gid >> 5) & 31;
    int nt   = gid >> 10;
    if (nt >= NT_PAD) return;
    int half = lane >> 4;
    int c    = nt * 16 + (lane & 15);
    unsigned short* base = Bws + ((((size_t)nt * KB + kb) * 32 + lane) * 32);
#pragma unroll
    for (int j = 0; j < 16; ++j) {
        int k = kb * 32 + j + 16 * half;        // K index in [0,1024)
        float val = 0.0f;
        if (c < N_COLS) {
            int f  = c >> 1;
            int pm = (f * k) & (NFFT - 1);      // exact range reduction of 2*pi*f*k/N
            float s, co;
            sincospif(2.0f * (float)pm * (1.0f / (float)NFFT), &s, &co);
            float wv = win[k];
            val = (c & 1) ? (-s * wv) : (co * wv);
        }
        unsigned short hi = f2bf(val);
        unsigned short lo = f2bf(val - bf2f(hi));
        base[j]      = hi;
        base[16 + j] = lo;
    }
}

// -----------------------------------------------------------------------------
// Kernel 2: GEMM  out[m, c] = sum_k frames[m, k] * basis[k, c]
// M = 30016 (32 rows per block), K = 1024, N padded to 1152.
// Split-bf16 (hi/lo) x3 WMMA per K=32 for fp32-grade accuracy on the bf16 pipe.
// Each wave holds 2 M-tiles x 9 N-tiles of accumulators (144 VGPRs), so each
// B fragment load feeds 6 WMMAs and each A fragment read feeds 27.
// Frames staged to LDS with on-the-fly reflect padding (never hit HBM).
// -----------------------------------------------------------------------------
__global__ __launch_bounds__(256) void stft_wmma_gemm(
        const float* __restrict__ x,
        const unsigned short* __restrict__ Bws,
        float* __restrict__ out) {
    __attribute__((aligned(16))) __shared__ unsigned short sA[2][M_PER_BLK][SROW]; // [hi/lo][row][k]

    const int mblk = blockIdx.x;                 // 0..937
    const int tid  = threadIdx.x;
    const int lane = tid & 31;
    const int wave = tid >> 5;
    const int half = lane >> 4;
    const int nlo  = lane & 15;

    v8f acc[2][9];
#pragma unroll
    for (int mt = 0; mt < 2; ++mt)
#pragma unroll
        for (int i = 0; i < 9; ++i)
            acc[mt][i] = (v8f){0.f,0.f,0.f,0.f,0.f,0.f,0.f,0.f};

    // staging mapping: 16 threads per row, two row-groups of 16 rows
    const int srow16 = tid >> 4;                 // 0..15
    const int scol0  = tid & 15;                 // + 16*j
    int          tt[2];
    const float* xb[2];
#pragma unroll
    for (int rr = 0; rr < 2; ++rr) {
        int m_st = mblk * M_PER_BLK + rr * 16 + srow16;
        int bb   = m_st / N_FRAMES;
        tt[rr]   = m_st - bb * N_FRAMES;
        xb[rr]   = x + (size_t)bb * T_LEN;
    }

    for (int kc = 0; kc < N_CHUNKS; ++kc) {
        __syncthreads();                          // previous chunk fully consumed
#pragma unroll
        for (int rr = 0; rr < 2; ++rr) {
            const int r = rr * 16 + srow16;
#pragma unroll 4
            for (int j = 0; j < 16; ++j) {        // stage 32 x 256 floats -> bf16 hi/lo
                int nloc = scol0 + 16 * j;
                int q    = tt[rr] * HOP + (kc * KCHUNK + nloc) - (NFFT / 2);
                int idx  = (q < 0) ? -q : ((q >= T_LEN) ? (2 * T_LEN - 2 - q) : q);
                float v  = xb[rr][idx];
                unsigned short hi = f2bf(v);
                unsigned short lo = f2bf(v - bf2f(hi));
                sA[0][r][nloc] = hi;
                sA[1][r][nloc] = lo;
            }
        }
        __syncthreads();

#pragma unroll 1
        for (int kbl = 0; kbl < KB_CHUNK; ++kbl) {
            const int kb   = kc * KB_CHUNK + kbl;
            const int koff = kbl * 32;
            // A fragments (bf16 16x32): row M = lane&15; K = j + 16*half
            V16U ahi[2], alo[2];
#pragma unroll
            for (int mt = 0; mt < 2; ++mt) {
                const unsigned short* ah = &sA[0][mt * 16 + nlo][koff + 16 * half];
                const unsigned short* al = &sA[1][mt * 16 + nlo][koff + 16 * half];
                ahi[mt].u[0] = *(const ush8*)(ah);
                ahi[mt].u[1] = *(const ush8*)(ah + 8);
                alo[mt].u[0] = *(const ush8*)(al);
                alo[mt].u[1] = *(const ush8*)(al + 8);
            }
#pragma unroll
            for (int i = 0; i < 9; ++i) {
                const int nt = wave + 8 * i;      // < 72 always (padded basis)
                const unsigned short* bp =
                    Bws + ((((size_t)nt * KB + kb) * 32 + lane) * 32);
                V16U bhi, blo;
                bhi.u[0] = ((const ush8*)bp)[0];
                bhi.u[1] = ((const ush8*)bp)[1];
                blo.u[0] = ((const ush8*)bp)[2];
                blo.u[1] = ((const ush8*)bp)[3];
#pragma unroll
                for (int mt = 0; mt < 2; ++mt) {
                    // split-fp32 emulation: hi*hi + hi*lo + lo*hi (lo*lo negligible)
                    acc[mt][i] = __builtin_amdgcn_wmma_f32_16x16x32_bf16(
                        false, ahi[mt].v, false, bhi.v, (short)0, acc[mt][i], false, false);
                    acc[mt][i] = __builtin_amdgcn_wmma_f32_16x16x32_bf16(
                        false, ahi[mt].v, false, blo.v, (short)0, acc[mt][i], false, false);
                    acc[mt][i] = __builtin_amdgcn_wmma_f32_16x16x32_bf16(
                        false, alo[mt].v, false, bhi.v, (short)0, acc[mt][i], false, false);
                }
            }
        }
    }

    // epilogue: D layout lane L: N = L&15, VGPR v -> M = v + 8*(L>>4)
    // non-temporal stores: keep the 123 MB output from evicting basis/x in L2
#pragma unroll
    for (int mt = 0; mt < 2; ++mt) {
#pragma unroll
        for (int i = 0; i < 9; ++i) {
            const int c = (wave + 8 * i) * 16 + nlo;
            if (c < N_COLS) {
#pragma unroll
                for (int v = 0; v < 8; ++v) {
                    const int m = mblk * M_PER_BLK + mt * 16 + v + 8 * half;
                    __builtin_nontemporal_store(acc[mt][i][v],
                                                &out[(size_t)m * N_COLS + c]);
                }
            }
        }
    }
}

extern "C" void kernel_launch(void* const* d_in, const int* in_sizes, int n_in,
                              void* d_out, int out_size, void* d_ws, size_t ws_size,
                              hipStream_t stream) {
    (void)in_sizes; (void)n_in; (void)out_size; (void)ws_size;
    const float* x   = (const float*)d_in[0];   // (16, 480000) f32
    const float* win = (const float*)d_in[1];   // (1024,) f32
    float* out = (float*)d_out;                 // (16, 1876, 513, 2) f32
    unsigned short* Bws = (unsigned short*)d_ws; // 72*32*32*32 bf16 = 4.72 MB

    // basis rebuilt every call (deterministic, no cross-call state)
    stft_build_basis<<<(NT_PAD * KB * 32) / 256, 256, 0, stream>>>(win, Bws);
    stft_wmma_gemm<<<N_BLOCKS, 256, 0, stream>>>(x, Bws, out);
}
